// BayesianLstm_63178968924712
// MI455X (gfx1250) — matrix operand order repaired and checked by reading, not compiled, not run on hardware
//
#include <hip/hip_runtime.h>
#include <hip/hip_bf16.h>
#include <math.h>

typedef __attribute__((ext_vector_type(16))) __bf16 v16bf;
typedef __attribute__((ext_vector_type(8)))  float  v8f;

#define B_TOT 8192
#define T_LEN 100
#define I_DIM 24
#define H1    64
#define H2    128
#define G1    256   // 4*H1
#define G2    512   // 4*H2

#define WMMA_BF16(a, b, c) \
  __builtin_amdgcn_wmma_f32_16x16x32_bf16(false, (a), false, (b), (short)0, (c), false, false)

// CDNA5 has a hardware tanh transcendental (v_tanh_f32); use it when the
// toolchain exposes the builtin, fall back to libm otherwise.
#if __has_builtin(__builtin_amdgcn_tanhf)
#define TANHF(x) __builtin_amdgcn_tanhf(x)
#else
#define TANHF(x) tanhf(x)
#endif

// sigmoid(x) = 0.5*tanh(x/2) + 0.5  -> no division, 1 trans op
__device__ __forceinline__ float sigmoidf_(float x) {
  return fmaf(0.5f, TANHF(0.5f * x), 0.5f);
}

// ---------------- weight sampling: w = mu + log1p(exp(rho)) * eps ----------------
__global__ void sample_bf16_pad(const float* __restrict__ mu, const float* __restrict__ rho,
                                const float* __restrict__ eps, __bf16* __restrict__ dst,
                                int srcRows, int dstRows, int cols) {
  int idx = blockIdx.x * blockDim.x + threadIdx.x;
  int total = dstRows * cols;
  if (idx >= total) return;
  int r = idx / cols, c = idx - r * cols;
  float v = 0.0f;
  if (r < srcRows) {
    int si = r * cols + c;
    v = mu[si] + log1pf(expf(rho[si])) * eps[si];
  }
  dst[idx] = (__bf16)v;
}

__global__ void sample_f32(const float* __restrict__ mu, const float* __restrict__ rho,
                           const float* __restrict__ eps, float* __restrict__ dst, int n) {
  int idx = blockIdx.x * blockDim.x + threadIdx.x;
  if (idx >= n) return;
  dst[idx] = mu[idx] + log1pf(expf(rho[idx])) * eps[idx];
}

// ---------------- WMMA fragment loaders (ISA 7.12.2 layouts, wave32) ----------------
// B fragment: 32x16 bf16 tile of row-major KxN matrix. lane -> N=lane&15;
// VGPR 0-3: K=0..7 (lanes 0-15) / K=8..15 (lanes 16-31); VGPR 4-7: +16.
__device__ __forceinline__ v16bf load_b_frag(const __bf16* __restrict__ W, int ldn,
                                             int kBase, int nBase, int lane) {
  int n  = nBase + (lane & 15);
  int kh = kBase + ((lane >> 4) << 3);
  v16bf f;
#pragma unroll
  for (int v = 0; v < 8; ++v) {
    int kb = kh + ((v < 4) ? (2 * v) : (16 + 2 * (v - 4)));
    f[2 * v]     = W[kb       * ldn + n];
    f[2 * v + 1] = W[(kb + 1) * ldn + n];
  }
  return f;
}

// A fragment: 16x32 bf16 tile of row-major MxK (LDS). lane -> M=lane&15;
// VGPR 0-3: K=0..7 (lanes 0-15) / K=8..15 (lanes 16-31); VGPR 4-7: +16.
__device__ __forceinline__ v16bf load_a_frag(const __bf16* __restrict__ A, int ldk,
                                             int kBase, int lane) {
  int m = lane & 15;
  int h = lane >> 4;
  v16bf f;
#pragma unroll
  for (int v = 0; v < 8; ++v) {
    int kb = kBase + ((v < 4) ? (2 * v + 8 * h) : (16 + 2 * (v - 4) + 8 * h));
    f[2 * v]     = A[m * ldk + kb];
    f[2 * v + 1] = A[m * ldk + kb + 1];
  }
  return f;
}

// ---------------- fused 2-layer LSTM + FC head ----------------
// grid = B/16 blocks of 512 threads (16 waves); each block owns a 16-row batch tile
// and runs the full T=100 scan with state resident in LDS, weights in registers.
__global__ void __launch_bounds__(512)
lstm_fused(const float* __restrict__ input,
           const __bf16* __restrict__ w1i, const __bf16* __restrict__ w1h,
           const __bf16* __restrict__ w2i, const __bf16* __restrict__ w2h,
           const float* __restrict__ b1, const float* __restrict__ b2,
           const float* __restrict__ fc1w, const float* __restrict__ fc1b,
           const float* __restrict__ fc2w, const float* __restrict__ fc2b,
           const float* __restrict__ outw, const float* __restrict__ outb,
           float* __restrict__ out) {
  __shared__ __bf16 xbuf[16 * 32];        // x_t tile, K padded 24->32
  __shared__ __bf16 h1buf[16 * H1];
  __shared__ float  c1buf[16 * H1];
  __shared__ __bf16 h2buf[16 * H2];
  __shared__ float  c2buf[16 * H2];
  __shared__ float  gates[16 * G2];       // shared between L1 (16x256) and L2 (16x512)

  const int tid  = threadIdx.x;
  const int lane = tid & 31;
  const int w    = tid >> 5;              // wave 0..15
  const int bat0 = blockIdx.x * 16;

  // per-thread staging slot for x: row = wave, col = lane (16 rows x 32 cols)
  const int srow = tid >> 5, scol = tid & 31;
  const float* xptr =
      input + (size_t)(bat0 + srow) * T_LEN * I_DIM + scol;   // + t*I_DIM per step

  // ---- preload loop-invariant weight B-fragments into registers ----
  const int n1 = 16 * w;                  // L1 gate-column base for this wave
  v16bf B1i  = load_b_frag(w1i, G1, 0,  n1, lane);
  v16bf B1h0 = load_b_frag(w1h, G1, 0,  n1, lane);
  v16bf B1h1 = load_b_frag(w1h, G1, 32, n1, lane);
  float bias1 = b1[n1 + (lane & 15)];

  v16bf B2i[2][2], B2h[2][4];
  float bias2[2];
#pragma unroll
  for (int tix = 0; tix < 2; ++tix) {
    int nb = 32 * w + 16 * tix;           // L2 gate-column base (2 tiles per wave)
    bias2[tix] = b2[nb + (lane & 15)];
#pragma unroll
    for (int kk = 0; kk < 2; ++kk) B2i[tix][kk] = load_b_frag(w2i, G2, 32 * kk, nb, lane);
#pragma unroll
    for (int kk = 0; kk < 4; ++kk) B2h[tix][kk] = load_b_frag(w2h, G2, 32 * kk, nb, lane);
  }

  // ---- zero state + stage x_0 ----
  for (int e = tid; e < 16 * H1; e += 512) { c1buf[e] = 0.0f; h1buf[e] = (__bf16)0.0f; }
  for (int e = tid; e < 16 * H2; e += 512) { c2buf[e] = 0.0f; h2buf[e] = (__bf16)0.0f; }
  {
    float v = (scol < I_DIM) ? xptr[0] : 0.0f;
    xbuf[srow * 32 + scol] = (__bf16)v;
  }
  __syncthreads();

  for (int t = 0; t < T_LEN; ++t) {
    // ---- layer 1 gates: x@W1i + h1@W1h + b1 (one 16x16 tile per wave) ----
    {
      v16bf Ax  = load_a_frag(xbuf, 32, 0, lane);
      v16bf Ah0 = load_a_frag(h1buf, H1, 0, lane);
      v16bf Ah1 = load_a_frag(h1buf, H1, 32, lane);
      v8f acc;
#pragma unroll
      for (int r = 0; r < 8; ++r) acc[r] = bias1;
      acc = WMMA_BF16(Ax,  B1i,  acc);
      acc = WMMA_BF16(Ah0, B1h0, acc);
      acc = WMMA_BF16(Ah1, B1h1, acc);
      int m0 = (lane >> 4) * 8, n = n1 + (lane & 15);
#pragma unroll
      for (int r = 0; r < 8; ++r) gates[(m0 + r) * G1 + n] = acc[r];
    }
    __syncthreads();

    // layer 1 cell update + stage x_{t+1} (xbuf's only reader ran above)
    for (int e = tid; e < 16 * H1; e += 512) {
      int row = e >> 6, j = e & (H1 - 1);
      const float* g = &gates[row * G1];
      float ig = sigmoidf_(g[j]);
      float fg = sigmoidf_(g[H1 + j]);
      float gg = TANHF(g[2 * H1 + j]);
      float og = sigmoidf_(g[3 * H1 + j]);
      float c = fg * c1buf[e] + ig * gg;
      c1buf[e] = c;
      h1buf[e] = (__bf16)(og * TANHF(c));
    }
    if (t + 1 < T_LEN) {
      float v = (scol < I_DIM) ? xptr[(t + 1) * I_DIM] : 0.0f;
      xbuf[srow * 32 + scol] = (__bf16)v;
    }
    __syncthreads();

    // ---- layer 2 gates: h1@W2i + h2@W2h + b2 (two 16x16 tiles per wave) ----
    {
      v16bf A1[2], A2[4];
#pragma unroll
      for (int kk = 0; kk < 2; ++kk) A1[kk] = load_a_frag(h1buf, H1, 32 * kk, lane);
#pragma unroll
      for (int kk = 0; kk < 4; ++kk) A2[kk] = load_a_frag(h2buf, H2, 32 * kk, lane);
#pragma unroll
      for (int tix = 0; tix < 2; ++tix) {
        v8f acc;
#pragma unroll
        for (int r = 0; r < 8; ++r) acc[r] = bias2[tix];
#pragma unroll
        for (int kk = 0; kk < 2; ++kk) acc = WMMA_BF16(A1[kk], B2i[tix][kk], acc);
#pragma unroll
        for (int kk = 0; kk < 4; ++kk) acc = WMMA_BF16(A2[kk], B2h[tix][kk], acc);
        int nb = 32 * w + 16 * tix;
        int m0 = (lane >> 4) * 8, n = nb + (lane & 15);
#pragma unroll
        for (int r = 0; r < 8; ++r) gates[(m0 + r) * G2 + n] = acc[r];
      }
    }
    __syncthreads();

    // layer 2 cell update
    for (int e = tid; e < 16 * H2; e += 512) {
      int row = e >> 7, j = e & (H2 - 1);
      const float* g = &gates[row * G2];
      float ig = sigmoidf_(g[j]);
      float fg = sigmoidf_(g[H2 + j]);
      float gg = TANHF(g[2 * H2 + j]);
      float og = sigmoidf_(g[3 * H2 + j]);
      float c = fg * c2buf[e] + ig * gg;
      c2buf[e] = c;
      h2buf[e] = (__bf16)(og * TANHF(c));
    }
    __syncthreads();
  }

  // ---- FC head on h2[:, -1, :]: relu(h2@fc1^T+b) -> relu(@fc2^T+b) -> @outw^T+b ----
  if (tid < 16) {
    const __bf16* h2 = &h2buf[tid * H2];
    float x1[8];
#pragma unroll
    for (int j = 0; j < 8; ++j) {
      float s = fc1b[j];
      for (int k = 0; k < H2; ++k) s += fc1w[j * H2 + k] * (float)h2[k];
      x1[j] = fmaxf(s, 0.0f);
    }
    float x2[8];
#pragma unroll
    for (int j = 0; j < 8; ++j) {
      float s = fc2b[j];
#pragma unroll
      for (int k = 0; k < 8; ++k) s += fc2w[j * 8 + k] * x1[k];
      x2[j] = fmaxf(s, 0.0f);
    }
    float y = outb[0];
#pragma unroll
    for (int j = 0; j < 8; ++j) y += outw[j] * x2[j];
    out[bat0 + tid] = y;
  }
}

extern "C" void kernel_launch(void* const* d_in, const int* in_sizes, int n_in,
                              void* d_out, int out_size, void* d_ws, size_t ws_size,
                              hipStream_t stream) {
  (void)in_sizes; (void)n_in; (void)out_size; (void)ws_size;
  const float* input = (const float*)d_in[0];
  auto P = [&](int i) { return (const float*)d_in[i]; };

  // workspace layout: bf16 weights first (even element counts), then f32 params
  __bf16* w1i = (__bf16*)d_ws;            // 32 x 256  (K padded 24->32)
  __bf16* w1h = w1i + 32 * G1;            // 64 x 256
  __bf16* w2i = w1h + H1 * G1;            // 64 x 512
  __bf16* w2h = w2i + H1 * G2;            // 128 x 512
  float*  b1   = (float*)(w2h + H2 * G2);
  float*  b2   = b1 + G1;
  float*  fc1w = b2 + G2;                 // 8 x 128
  float*  fc1b = fc1w + 8 * H2;
  float*  fc2w = fc1b + 8;                // 8 x 8
  float*  fc2b = fc2w + 64;
  float*  outw = fc2b + 8;                // 1 x 8
  float*  outb = outw + 8;

  auto gsz = [](int n) { return (n + 255) / 256; };
  sample_bf16_pad<<<gsz(32 * G1), 256, 0, stream>>>(P(1),  P(2),  P(3),  w1i, I_DIM, 32, G1);
  sample_bf16_pad<<<gsz(H1 * G1), 256, 0, stream>>>(P(4),  P(5),  P(6),  w1h, H1, H1, G1);
  sample_bf16_pad<<<gsz(H1 * G2), 256, 0, stream>>>(P(10), P(11), P(12), w2i, H1, H1, G2);
  sample_bf16_pad<<<gsz(H2 * G2), 256, 0, stream>>>(P(13), P(14), P(15), w2h, H2, H2, G2);
  sample_f32<<<gsz(G1),     256, 0, stream>>>(P(7),  P(8),  P(9),  b1,   G1);
  sample_f32<<<gsz(G2),     256, 0, stream>>>(P(16), P(17), P(18), b2,   G2);
  sample_f32<<<gsz(8 * H2), 256, 0, stream>>>(P(19), P(20), P(21), fc1w, 8 * H2);
  sample_f32<<<1, 32, 0, stream>>>(P(22), P(23), P(24), fc1b, 8);
  sample_f32<<<1, 64, 0, stream>>>(P(25), P(26), P(27), fc2w, 64);
  sample_f32<<<1, 32, 0, stream>>>(P(28), P(29), P(30), fc2b, 8);
  sample_f32<<<1, 32, 0, stream>>>(P(31), P(32), P(33), outw, 8);
  sample_f32<<<1, 32, 0, stream>>>(P(34), P(35), P(36), outb, 1);

  lstm_fused<<<B_TOT / 16, 512, 0, stream>>>(input, w1i, w1h, w2i, w2h, b1, b2,
                                             fc1w, fc1b, fc2w, fc2b, outw, outb,
                                             (float*)d_out);
}
